// DCRNNModel_nextTimePred_10273561772737
// MI455X (gfx1250) — matrix operand order repaired and checked by reading, not compiled, not run on hardware
//
#include <hip/hip_runtime.h>
#include <math.h>

// ---------------------------------------------------------------------------
// DCRNN (diffusion-conv GRU seq2seq) for MI455X / gfx1250.
// GEMMs use v_wmma_f32_16x16x32_bf16 (bf16 inputs, f32 accumulate).
// ---------------------------------------------------------------------------

typedef __attribute__((ext_vector_type(16))) __bf16 v16bf;
typedef __attribute__((ext_vector_type(8)))  float  v8f;

constexpr int Bc   = 256;   // batch
constexpr int Tc   = 64;    // time steps (in == out)
constexpr int Nc   = 19;    // graph nodes
constexpr int DINc = 100;   // encoder input feat
constexpr int Uc   = 128;   // hidden units
constexpr int Oc   = 100;   // output feat
constexpr int ROWS = Bc * Nc;        // 4864 (multiple of 16)
constexpr int GDIM = 2 * Uc;         // 256 gate outputs
constexpr int K0   = (DINc + Uc) * 3;   // 684
constexpr int K0P  = 704;               // padded to multiple of 32
constexpr int K1   = (Uc + Uc) * 3;     // 768
constexpr int K1P  = 768;
constexpr int KPRJ = 128;               // projection K (already mult of 32)
constexpr int OPAD = 112;               // projection out padded to mult of 16

__device__ __forceinline__ unsigned short f2bf(float f) {
  unsigned int u = __float_as_uint(f);
  u += 0x7fffu + ((u >> 16) & 1u);   // round-to-nearest-even
  return (unsigned short)(u >> 16);
}

// ---------------------------------------------------------------------------
// Fused: concat(x, h or r*h) -> Chebyshev diffusion (x0, S x0, 2 S x1 - x0)
// -> bf16 X with torch matrix-minor layout (col = d*3 + matIdx), zero-padded
// to Kpad columns.  One block per batch element; S (19x19) staged in LDS.
// ---------------------------------------------------------------------------
template<bool USE_R>
__global__ void build_X_kernel(const float* __restrict__ x, int xBStride, int xMStride, int xF,
                               const float* __restrict__ h,
                               const float* __restrict__ r, int rStride,
                               const float* __restrict__ S,
                               int D, int Kpad,
                               unsigned short* __restrict__ X) {
  __shared__ float sS[Nc * Nc];
  const int b = blockIdx.x;
  for (int i = threadIdx.x; i < Nc * Nc; i += blockDim.x) sS[i] = S[i];
  __syncthreads();
  const int D3 = 3 * D;
  for (int d = threadIdx.x; d < D; d += blockDim.x) {
    float xv[Nc], x1[Nc];
    if (d < xF) {
#pragma unroll
      for (int m = 0; m < Nc; ++m) xv[m] = x[(size_t)b * xBStride + m * xMStride + d];
    } else {
      const int dh = d - xF;
#pragma unroll
      for (int m = 0; m < Nc; ++m) {
        float hv = h[(size_t)(b * Nc + m) * Uc + dh];
        if (USE_R) hv *= r[(size_t)(b * Nc + m) * rStride + dh];
        xv[m] = hv;
      }
    }
#pragma unroll
    for (int n = 0; n < Nc; ++n) {
      float s = 0.f;
#pragma unroll
      for (int m = 0; m < Nc; ++m) s += sS[n * Nc + m] * xv[m];
      x1[n] = s;
    }
#pragma unroll
    for (int n = 0; n < Nc; ++n) {
      float s = 0.f;
#pragma unroll
      for (int m = 0; m < Nc; ++m) s += sS[n * Nc + m] * x1[m];
      const float x2 = 2.f * s - xv[n];
      unsigned short* p = X + (size_t)(b * Nc + n) * Kpad + 3 * d;
      p[0] = f2bf(xv[n]);
      p[1] = f2bf(x1[n]);
      p[2] = f2bf(x2);
    }
  }
  // zero K padding (684 -> 704 for layer-0 shapes; no-op when D3 == Kpad)
  for (int c = D3 + threadIdx.x; c < Kpad; c += blockDim.x)
    for (int n = 0; n < Nc; ++n)
      X[(size_t)(b * Nc + n) * Kpad + c] = 0;
}

// ---------------------------------------------------------------------------
// WMMA GEMM: out[ROWS x outDim] = act(X[ROWS x Kpad] * Wt^T + bias)
// Wt is outPad x Kpad row-major (i.e., W transposed, zero-padded).
// 4 waves per block, one 16x16 D-tile per wave, K-loop step 32 (bf16).
// Optional second (strided) destination for the decoder projection.
// ACT: 0=none, 1=sigmoid, 2=tanh.
// ---------------------------------------------------------------------------
union Frag { v16bf v; uint4 q[2]; };

template<int ACT>
__global__ void wmma_gemm_kernel(const unsigned short* __restrict__ X, int Kpad,
                                 const unsigned short* __restrict__ Wt,
                                 const float* __restrict__ bias,
                                 int outDim, int outPad,
                                 float* __restrict__ out, int ldOut,
                                 float* __restrict__ out2, int tstep) {
  const int lane = threadIdx.x & 31;
  const int wv   = threadIdx.x >> 5;
  const int rowTile = blockIdx.x;
  const int colTile = blockIdx.y * 4 + wv;
  if (colTile * 16 >= outPad) return;

  const int m16 = lane & 15;   // A row / B column within tile
  const int kh  = lane >> 4;   // lane half selects K sub-range
  const uint4* __restrict__ Arow = (const uint4*)(X  + (size_t)(rowTile * 16 + m16) * Kpad);
  const uint4* __restrict__ Brow = (const uint4*)(Wt + (size_t)(colTile * 16 + m16) * Kpad);

  v8f acc = (v8f)(0.0f);
  for (int k0 = 0; k0 < Kpad; k0 += 32) {
    Frag a, bf;
    // A 16x32 bf16 layout: lanes 0-15 K={0..7,16..23}, lanes 16-31 K={8..15,24..31}
    const int ia = (k0 >> 3) + kh;
    a.q[0] = Arow[ia];
    a.q[1] = Arow[ia + 2];
    // B 32x16 bf16 layout: lanes 0-15 K={0..7,16..23}, lanes 16-31 K={8..15,24..31}
    const int ib = (k0 >> 3) + kh;
    bf.q[0] = Brow[ib];
    bf.q[1] = Brow[ib + 2];
    acc = __builtin_amdgcn_wmma_f32_16x16x32_bf16(false, a.v, false, bf.v,
                                                  (short)0, acc, false, false);
  }

  // C/D layout: VGPR i -> M = i + 8*(lane>=16), N = lane%16
  const int col = colTile * 16 + (lane & 15);
  if (col >= outDim) return;
  const float bv = bias[col];
  const int moff = (lane >> 4) << 3;
#pragma unroll
  for (int i = 0; i < 8; ++i) {
    const int row = rowTile * 16 + i + moff;
    float v = acc[i] + bv;
    if (ACT == 1)      v = 1.f / (1.f + expf(-v));
    else if (ACT == 2) v = tanhf(v);
    out[(size_t)row * ldOut + col] = v;
    if (out2) {  // projection also scatters to (B, T, N, O) output tensor
      const int b = row / Nc, n = row % Nc;
      out2[(((size_t)b * Tc + tstep) * Nc + n) * Oc + col] = v;
    }
  }
}

// ---------------------------------------------------------------------------
// Support kernels.
// ---------------------------------------------------------------------------
__global__ void conv_wt_kernel(const float* __restrict__ W, int K, int outDim,
                               int Kpad, unsigned short* __restrict__ Wt) {
  const int o = blockIdx.x;  // grid.x == outPad
  for (int k = threadIdx.x; k < Kpad; k += blockDim.x) {
    unsigned short v = 0;
    if (o < outDim && k < K) v = f2bf(W[(size_t)k * outDim + o]);
    Wt[(size_t)o * Kpad + k] = v;
  }
}

__global__ void update_h_kernel(const float* __restrict__ gates,
                                const float* __restrict__ c,
                                float* __restrict__ h) {
  const int idx = blockIdx.x * blockDim.x + threadIdx.x;
  if (idx >= ROWS * Uc) return;
  const int row = idx / Uc, col = idx % Uc;
  const float u = gates[(size_t)row * GDIM + Uc + col];
  h[idx] = u * h[idx] + (1.f - u) * c[idx];
}

__global__ void f32_to_bf16_kernel(const float* __restrict__ a,
                                   unsigned short* __restrict__ o, int n) {
  const int idx = blockIdx.x * blockDim.x + threadIdx.x;
  if (idx < n) o[idx] = f2bf(a[idx]);
}

__global__ void zero_f32_kernel(float* __restrict__ p, int n) {
  const int idx = blockIdx.x * blockDim.x + threadIdx.x;
  if (idx < n) p[idx] = 0.f;
}

// ---------------------------------------------------------------------------
// Host side.
// ---------------------------------------------------------------------------
namespace {

struct CellW {
  const unsigned short* Wg; const float* bg;
  const unsigned short* Wc; const float* bc;
  int Kpad;
};

void launch_gemm(int act, const unsigned short* X, int Kpad,
                 const unsigned short* Wt, const float* bias,
                 int outDim, int outPad, float* out, int ldOut,
                 float* out2, int tstep, hipStream_t s) {
  dim3 grid(ROWS / 16, (outPad / 16 + 3) / 4);
  dim3 block(128);
  switch (act) {
    case 1: wmma_gemm_kernel<1><<<grid, block, 0, s>>>(X, Kpad, Wt, bias, outDim, outPad, out, ldOut, out2, tstep); break;
    case 2: wmma_gemm_kernel<2><<<grid, block, 0, s>>>(X, Kpad, Wt, bias, outDim, outPad, out, ldOut, out2, tstep); break;
    default: wmma_gemm_kernel<0><<<grid, block, 0, s>>>(X, Kpad, Wt, bias, outDim, outPad, out, ldOut, out2, tstep); break;
  }
}

void run_cell(const float* x, int xB, int xM, int xF,
              float* h, const CellW& w,
              unsigned short* Xbuf, float* gates, float* cbuf,
              const float* S, hipStream_t s) {
  const int D = xF + Uc;
  build_X_kernel<false><<<Bc, 256, 0, s>>>(x, xB, xM, xF, h, nullptr, 0, S, D, w.Kpad, Xbuf);
  launch_gemm(/*sigmoid*/1, Xbuf, w.Kpad, w.Wg, w.bg, GDIM, GDIM, gates, GDIM, nullptr, 0, s);
  build_X_kernel<true><<<Bc, 256, 0, s>>>(x, xB, xM, xF, h, gates, GDIM, S, D, w.Kpad, Xbuf);
  launch_gemm(/*tanh*/2, Xbuf, w.Kpad, w.Wc, w.bc, Uc, Uc, cbuf, Uc, nullptr, 0, s);
  update_h_kernel<<<(ROWS * Uc + 255) / 256, 256, 0, s>>>(gates, cbuf, h);
}

} // namespace

extern "C" void kernel_launch(void* const* d_in, const int* in_sizes, int n_in,
                              void* d_out, int out_size, void* d_ws, size_t ws_size,
                              hipStream_t stream) {
  (void)in_sizes; (void)n_in; (void)out_size; (void)ws_size;
  const float* enc   = (const float*)d_in[0];
  /* d_in[1] = decoder_inputs: unused (autoregressive feedback) */
  const float* S     = (const float*)d_in[2];
  const float* e0_Wg = (const float*)d_in[3];  const float* e0_bg = (const float*)d_in[4];
  const float* e0_Wc = (const float*)d_in[5];  const float* e0_bc = (const float*)d_in[6];
  const float* e1_Wg = (const float*)d_in[7];  const float* e1_bg = (const float*)d_in[8];
  const float* e1_Wc = (const float*)d_in[9];  const float* e1_bc = (const float*)d_in[10];
  const float* d0_Wg = (const float*)d_in[11]; const float* d0_bg = (const float*)d_in[12];
  const float* d0_Wc = (const float*)d_in[13]; const float* d0_bc = (const float*)d_in[14];
  const float* d1_Wg = (const float*)d_in[15]; const float* d1_bg = (const float*)d_in[16];
  const float* d1_Wc = (const float*)d_in[17]; const float* d1_bc = (const float*)d_in[18];
  const float* Wp    = (const float*)d_in[19]; const float* bp    = (const float*)d_in[20];
  float* out = (float*)d_out;

  // --- carve workspace (256-byte aligned slices) ---
  char*  ws  = (char*)d_ws;
  size_t off = 0;
  auto alloc = [&](size_t bytes) -> void* {
    void* p = ws + off;
    off = (off + bytes + 255) & ~(size_t)255;
    return p;
  };
  unsigned short* Xbuf  = (unsigned short*)alloc((size_t)ROWS * K1P * 2); // max Kpad
  float*          gates = (float*)alloc((size_t)ROWS * GDIM * 4);
  float*          cbuf  = (float*)alloc((size_t)ROWS * Uc * 4);
  float*          h0    = (float*)alloc((size_t)ROWS * Uc * 4);
  float*          h1    = (float*)alloc((size_t)ROWS * Uc * 4);
  float*          cur   = (float*)alloc((size_t)ROWS * Oc * 4);
  unsigned short* w_e0g = (unsigned short*)alloc((size_t)GDIM * K0P * 2);
  unsigned short* w_e0c = (unsigned short*)alloc((size_t)Uc   * K0P * 2);
  unsigned short* w_e1g = (unsigned short*)alloc((size_t)GDIM * K1P * 2);
  unsigned short* w_e1c = (unsigned short*)alloc((size_t)Uc   * K1P * 2);
  unsigned short* w_d0g = (unsigned short*)alloc((size_t)GDIM * K0P * 2);
  unsigned short* w_d0c = (unsigned short*)alloc((size_t)Uc   * K0P * 2);
  unsigned short* w_d1g = (unsigned short*)alloc((size_t)GDIM * K1P * 2);
  unsigned short* w_d1c = (unsigned short*)alloc((size_t)Uc   * K1P * 2);
  unsigned short* w_p   = (unsigned short*)alloc((size_t)OPAD * KPRJ * 2);

  // --- weights -> bf16, transposed (out x Kpad), zero-padded ---
  conv_wt_kernel<<<GDIM, 128, 0, stream>>>(e0_Wg, K0, GDIM, K0P, w_e0g);
  conv_wt_kernel<<<Uc,   128, 0, stream>>>(e0_Wc, K0, Uc,   K0P, w_e0c);
  conv_wt_kernel<<<GDIM, 128, 0, stream>>>(e1_Wg, K1, GDIM, K1P, w_e1g);
  conv_wt_kernel<<<Uc,   128, 0, stream>>>(e1_Wc, K1, Uc,   K1P, w_e1c);
  conv_wt_kernel<<<GDIM, 128, 0, stream>>>(d0_Wg, K0, GDIM, K0P, w_d0g);
  conv_wt_kernel<<<Uc,   128, 0, stream>>>(d0_Wc, K0, Uc,   K0P, w_d0c);
  conv_wt_kernel<<<GDIM, 128, 0, stream>>>(d1_Wg, K1, GDIM, K1P, w_d1g);
  conv_wt_kernel<<<Uc,   128, 0, stream>>>(d1_Wc, K1, Uc,   K1P, w_d1c);
  conv_wt_kernel<<<OPAD, 128, 0, stream>>>(Wp,    Uc, Oc,   KPRJ, w_p);

  const CellW ce0{w_e0g, e0_bg, w_e0c, e0_bc, K0P};
  const CellW ce1{w_e1g, e1_bg, w_e1c, e1_bc, K1P};
  const CellW cd0{w_d0g, d0_bg, w_d0c, d0_bc, K0P};
  const CellW cd1{w_d1g, d1_bg, w_d1c, d1_bc, K1P};

  const int nHU = ROWS * Uc;
  zero_f32_kernel<<<(nHU + 255) / 256, 256, 0, stream>>>(h0, nHU);
  zero_f32_kernel<<<(nHU + 255) / 256, 256, 0, stream>>>(h1, nHU);

  // --- encoder: 2 stacked DCGRU layers, interleaved per time step ---
  for (int t = 0; t < Tc; ++t) {
    const float* xt = enc + (size_t)t * Nc * DINc;   // (b,t,n,d): b-stride = T*N*Din
    run_cell(xt, Tc * Nc * DINc, DINc, DINc, h0, ce0, Xbuf, gates, cbuf, S, stream);
    run_cell(h0, Nc * Uc,        Uc,   Uc,   h1, ce1, Xbuf, gates, cbuf, S, stream);
  }

  // --- decoder: autoregressive, feeds projection back ---
  const int nCur = ROWS * Oc;
  zero_f32_kernel<<<(nCur + 255) / 256, 256, 0, stream>>>(cur, nCur);  // go symbol
  for (int t = 0; t < Tc; ++t) {
    run_cell(cur, Nc * Oc, Oc, Oc, h0, cd0, Xbuf, gates, cbuf, S, stream);
    run_cell(h0,  Nc * Uc, Uc, Uc, h1, cd1, Xbuf, gates, cbuf, S, stream);
    // projection: bf16(h1) @ Wp + bp -> cur (feedback) and d_out[:, t, :, :]
    f32_to_bf16_kernel<<<(nHU + 255) / 256, 256, 0, stream>>>(h1, Xbuf, nHU);
    launch_gemm(/*none*/0, Xbuf, KPRJ, w_p, bp, Oc, OPAD, cur, Oc, out, t, stream);
  }
}
